// MultiheadSelfAttentionWithBias_14147622273428
// MI455X (gfx1250) — matrix operand order repaired and checked by reading, not compiled, not run on hardware
//
#include <hip/hip_runtime.h>
#include <hip/hip_bf16.h>

// ---------------------------------------------------------------------------
// MultiheadSelfAttentionWithBias for MI455X (gfx1250), wave32 + WMMA bf16
// + async global->LDS staging (ASYNCcnt) for weight / K / V tiles.
// B=2, N=2048, D=1024, H=16, DH=64. f32 in/out, bf16 WMMA, f32 accumulate.
// ---------------------------------------------------------------------------

typedef __attribute__((ext_vector_type(16))) __bf16 v16bf;
typedef __attribute__((ext_vector_type(8)))  __bf16 v8bf;
typedef __attribute__((ext_vector_type(8)))  float  v8f;

#define B_   2
#define N_   2048
#define D_   1024
#define H_   16
#define DH_  64
#define SCALE_ 0.125f   // 1/sqrt(64)

union BF16x16 { v8bf h[2]; v16bf v; };

// Async copy 16B global -> LDS (gfx1250). lds_addr = LDS byte address
// (low 32 bits of the generic shared-aperture address, ISA 10.2).
#define ASYNC_LD_B128(lds_u32, gaddr_u64)                                  \
    asm volatile("global_load_async_to_lds_b128 %0, %1, off"               \
                 :: "v"(lds_u32), "v"(gaddr_u64) : "memory")

#define WAIT_ASYNC(n)                                                      \
    asm volatile("s_wait_asynccnt %0" :: "n"(n) : "memory")

static __device__ __forceinline__ unsigned lds_addr_of(const void* p) {
    return (unsigned)(unsigned long long)(uintptr_t)p;
}

// ---------------------------------------------------------------------------
// f32 -> bf16 conversion (grid-stride)
// ---------------------------------------------------------------------------
__global__ void cvt_f32_bf16(const float* __restrict__ src,
                             __bf16* __restrict__ dst, int n) {
    int i = blockIdx.x * blockDim.x + threadIdx.x;
    int stride = gridDim.x * blockDim.x;
    for (; i < n; i += stride) dst[i] = (__bf16)src[i];
}

// ---------------------------------------------------------------------------
// Shared GEMM core pattern: block = 4 waves (128 thr), tile M=128 x N=64.
// Weight B tile (64 cols x 32 k, bf16, [col][k]) staged in LDS via async
// loads, double buffered; each wave computes its own 32-row M slice.
// ---------------------------------------------------------------------------
static __device__ __forceinline__ void issue_b_tile(const __bf16* __restrict__ wsrc,
                                                    int nbase, int k0,
                                                    __bf16* sBuf /*shared*/) {
    // 64 cols * 32 k * 2B = 4KB = 256 x 16B chunks; 128 threads -> 2 each.
#pragma unroll
    for (int c = 0; c < 2; ++c) {
        int ch  = threadIdx.x * 2 + c;
        int col = ch >> 2;
        int kc  = (ch & 3) * 8;
        const __bf16* g = wsrc + (size_t)(nbase + col) * D_ + k0 + kc;
        ASYNC_LD_B128(lds_addr_of(&sBuf[col * 32 + kc]),
                      (unsigned long long)(uintptr_t)g);
    }
}

// ---------------------------------------------------------------------------
// QKV GEMM: qkv = x @ Wqkv^T, scattered into Q (x SCALE), K, V^T buffers.
// ---------------------------------------------------------------------------
__global__ __launch_bounds__(128) void qkv_gemm(const __bf16* __restrict__ xb,
                                                const __bf16* __restrict__ wq,
                                                __bf16* __restrict__ qb,
                                                __bf16* __restrict__ kb,
                                                __bf16* __restrict__ vt) {
    __shared__ __align__(16) __bf16 sB[2][64 * 32];

    const int lane  = threadIdx.x & 31;
    const int wave  = threadIdx.x >> 5;
    const int l15   = lane & 15;
    const int lhalf = (lane >> 4) & 1;
    const int mbase = blockIdx.x * 128 + wave * 32;
    const int nbase = blockIdx.y * 64;

    v8f acc[2][4] = {};

    issue_b_tile(wq, nbase, 0, sB[0]);

    for (int k0 = 0; k0 < D_; k0 += 32) {
        const int cur = (k0 >> 5) & 1;
        if (k0 + 32 < D_) {
            issue_b_tile(wq, nbase, k0 + 32, sB[cur ^ 1]);
            WAIT_ASYNC(2);     // our 2 chunks of the current tile are done
        } else {
            WAIT_ASYNC(0);
        }
        __syncthreads();       // all waves' chunks of current tile visible

        BF16x16 a[2];
#pragma unroll
        for (int mi = 0; mi < 2; ++mi) {
            const __bf16* ap = xb + (size_t)(mbase + mi * 16 + l15) * D_ + k0 + lhalf * 16;
            a[mi].h[0] = *(const v8bf*)(ap);
            a[mi].h[1] = *(const v8bf*)(ap + 8);
            if (k0 + 32 < D_) __builtin_prefetch(ap + 32, 0, 1);
        }
#pragma unroll
        for (int ni = 0; ni < 4; ++ni) {
            v16bf b = *(const v16bf*)&sB[cur][(ni * 16 + l15) * 32 + lhalf * 16];
#pragma unroll
            for (int mi = 0; mi < 2; ++mi)
                acc[mi][ni] = __builtin_amdgcn_wmma_f32_16x16x32_bf16(
                    false, a[mi].v, false, b, (short)0, acc[mi][ni], false, false);
        }
        __syncthreads();       // done reading current tile before it is reused
    }

    // Scatter: C layout = row (i + lhalf*8), col l15
#pragma unroll
    for (int mi = 0; mi < 2; ++mi)
#pragma unroll
    for (int ni = 0; ni < 4; ++ni)
#pragma unroll
    for (int i = 0; i < 8; ++i) {
        int m   = mbase + mi * 16 + i + lhalf * 8;
        int gn  = nbase + ni * 16 + l15;
        int bi  = m >> 11, tok = m & (N_ - 1);
        int sel = gn >> 10, rem = gn & (D_ - 1);
        int h   = rem >> 6, dh = rem & (DH_ - 1);
        float v = acc[mi][ni][i];
        size_t idx_qk = ((size_t)(bi * H_ + h) * N_ + tok) * DH_ + dh;
        if (sel == 0)      qb[idx_qk] = (__bf16)(v * SCALE_);
        else if (sel == 1) kb[idx_qk] = (__bf16)v;
        else               vt[((size_t)(bi * H_ + h) * DH_ + dh) * N_ + tok] = (__bf16)v;
    }
}

// ---------------------------------------------------------------------------
// Flash attention: one wave per (b, h, 16-query-row tile). Online softmax
// over 32-key chunks. K and V^T tiles async-staged into LDS, double
// buffered (prefetch depth 1). Scale is pre-folded into Q.
// ---------------------------------------------------------------------------
__global__ __launch_bounds__(32) void flash_attn(const __bf16* __restrict__ qb,
                                                 const __bf16* __restrict__ kb,
                                                 const __bf16* __restrict__ vt,
                                                 const float* __restrict__ bias,
                                                 const unsigned char* __restrict__ kpm,
                                                 __bf16* __restrict__ yw) {
    __shared__ __align__(32) float  sS[16 * 32];
    __shared__ __align__(32) float  sRow[16];
    __shared__ __align__(16) __bf16 sP[16 * 32];
    __shared__ __align__(16) __bf16 sK[2][32 * 64];   // [key][dh]
    __shared__ __align__(16) __bf16 sV[2][64 * 32];   // [dh][key]

    const int lane  = threadIdx.x;
    const int l15   = lane & 15;
    const int lhalf = (lane >> 4) & 1;
    const int mt    = blockIdx.x & (N_ / 16 - 1);   // 128 row-tiles
    const int bh    = blockIdx.x >> 7;
    const int b     = bh >> 4;
    const int h     = bh & (H_ - 1);
    const int mbase = mt * 16;

    const __bf16* kbase_p = kb + (size_t)bh * N_ * DH_;
    const __bf16* vbase_p = vt + (size_t)bh * DH_ * N_;

    // Issue async staging of one 32-key chunk (K: 32x64, V^T: 64x32) = 8+8
    // b128 chunks per lane.
    auto issue_kv = [&](int j0, int buf) {
#pragma unroll
        for (int c = 0; c < 8; ++c) {               // K tile
            int ch  = c * 32 + lane;                // 256 chunks
            int key = ch >> 3;
            int dhc = (ch & 7) * 8;
            const __bf16* g = kbase_p + (size_t)(j0 + key) * DH_ + dhc;
            ASYNC_LD_B128(lds_addr_of(&sK[buf][key * 64 + dhc]),
                          (unsigned long long)(uintptr_t)g);
        }
#pragma unroll
        for (int c = 0; c < 8; ++c) {               // V^T tile
            int ch = c * 32 + lane;
            int dh = ch >> 2;
            int kc = (ch & 3) * 8;
            const __bf16* g = vbase_p + (size_t)dh * N_ + j0 + kc;
            ASYNC_LD_B128(lds_addr_of(&sV[buf][dh * 32 + kc]),
                          (unsigned long long)(uintptr_t)g);
        }
    };

    // Preload Q A-frags (DH=64 => two 16x32 k-steps)
    const __bf16* qrow = qb + ((size_t)bh * N_ + mbase + l15) * DH_;
    BF16x16 aq[2];
#pragma unroll
    for (int ks = 0; ks < 2; ++ks) {
        aq[ks].h[0] = *(const v8bf*)(qrow + ks * 32 + lhalf * 16);
        aq[ks].h[1] = *(const v8bf*)(qrow + ks * 32 + lhalf * 16 + 8);
    }

    v8f o[4] = {};
    float mrow = -1e30f, lrow = 0.0f;
    const float* bias_row = bias + ((size_t)b * N_ + (mbase + l15)) * N_;
    const unsigned char* kpb = kpm + (size_t)b * N_;

    issue_kv(0, 0);

    for (int j0 = 0; j0 < N_; j0 += 32) {
        const int cur = (j0 >> 5) & 1;
        if (j0 + 32 < N_) {
            issue_kv(j0 + 32, cur ^ 1);
            WAIT_ASYNC(16);     // oldest 16 (current tile) complete, in-order
        } else {
            WAIT_ASYNC(0);
        }

        // S tile (16 rows x 32 keys) = Q @ K^T from LDS, staged into sS
#pragma unroll
        for (int jt = 0; jt < 2; ++jt) {
            v8f c = {};
#pragma unroll
            for (int ks = 0; ks < 2; ++ks) {
                v16bf bk = *(const v16bf*)
                    &sK[cur][(jt * 16 + l15) * 64 + ks * 32 + lhalf * 16];
                c = __builtin_amdgcn_wmma_f32_16x16x32_bf16(
                        false, aq[ks].v, false, bk, (short)0, c, false, false);
            }
#pragma unroll
            for (int i = 0; i < 8; ++i)
                sS[(i + lhalf * 8) * 32 + jt * 16 + l15] = c[i];
        }
        __syncthreads();

        // Per-row online softmax: lane r owns query row r (lanes 0..15)
        if (lane < 16) {
            float s[32];
            float mx = mrow;
#pragma unroll
            for (int c2 = 0; c2 < 32; ++c2) {
                float v = sS[lane * 32 + c2] + bias_row[j0 + c2];
                v = kpb[j0 + c2] ? v : -1e30f;
                s[c2] = v;
                mx = fmaxf(mx, v);
            }
            float corr = __expf(mrow - mx);
            float lsum = lrow * corr;
#pragma unroll
            for (int c2 = 0; c2 < 32; ++c2) {
                float p = __expf(s[c2] - mx);
                lsum += p;
                sP[lane * 32 + c2] = (__bf16)p;
            }
            mrow = mx;
            lrow = lsum;
            sRow[lane] = corr;
        }
        __syncthreads();

        // Rescale O accumulators by per-row correction (rows i+lhalf*8)
        v8f corrv = *(const v8f*)&sRow[lhalf * 8];
#pragma unroll
        for (int t = 0; t < 4; ++t) o[t] *= corrv;

        // P A-frag from LDS, then O += P @ V (V^T tile in LDS)
        BF16x16 pf;
        pf.h[0] = *(const v8bf*)&sP[l15 * 32 + lhalf * 16];
        pf.h[1] = *(const v8bf*)&sP[l15 * 32 + lhalf * 16 + 8];
#pragma unroll
        for (int t = 0; t < 4; ++t) {
            v16bf bv = *(const v16bf*)
                &sV[cur][(t * 16 + l15) * 32 + lhalf * 16];
            o[t] = __builtin_amdgcn_wmma_f32_16x16x32_bf16(
                       false, pf.v, false, bv, (short)0, o[t], false, false);
        }
        __syncthreads();   // protect sS/sP reuse next iteration
    }

    // Final 1/rowsum normalization, write bf16 [B, N, H*DH]
    if (lane < 16) sRow[lane] = 1.0f / lrow;
    __syncthreads();
    v8f rs = *(const v8f*)&sRow[lhalf * 8];
#pragma unroll
    for (int t = 0; t < 4; ++t)
#pragma unroll
    for (int i = 0; i < 8; ++i) {
        int tok = mbase + i + lhalf * 8;
        int dh  = t * 16 + l15;
        yw[((size_t)b * N_ + tok) * D_ + h * DH_ + dh] = (__bf16)(o[t][i] * rs[i]);
    }
}

// ---------------------------------------------------------------------------
// Output projection: out = yw @ Wproj^T + bproj  (f32 output)
// ---------------------------------------------------------------------------
__global__ __launch_bounds__(128) void proj_gemm(const __bf16* __restrict__ yw,
                                                 const __bf16* __restrict__ wp,
                                                 const float* __restrict__ bproj,
                                                 float* __restrict__ out) {
    __shared__ __align__(16) __bf16 sB[2][64 * 32];

    const int lane  = threadIdx.x & 31;
    const int wave  = threadIdx.x >> 5;
    const int l15   = lane & 15;
    const int lhalf = (lane >> 4) & 1;
    const int mbase = blockIdx.x * 128 + wave * 32;
    const int nbase = blockIdx.y * 64;

    v8f acc[2][4] = {};

    issue_b_tile(wp, nbase, 0, sB[0]);

    for (int k0 = 0; k0 < D_; k0 += 32) {
        const int cur = (k0 >> 5) & 1;
        if (k0 + 32 < D_) {
            issue_b_tile(wp, nbase, k0 + 32, sB[cur ^ 1]);
            WAIT_ASYNC(2);
        } else {
            WAIT_ASYNC(0);
        }
        __syncthreads();

        BF16x16 a[2];
#pragma unroll
        for (int mi = 0; mi < 2; ++mi) {
            const __bf16* ap = yw + (size_t)(mbase + mi * 16 + l15) * D_ + k0 + lhalf * 16;
            a[mi].h[0] = *(const v8bf*)(ap);
            a[mi].h[1] = *(const v8bf*)(ap + 8);
            if (k0 + 32 < D_) __builtin_prefetch(ap + 32, 0, 1);
        }
#pragma unroll
        for (int ni = 0; ni < 4; ++ni) {
            v16bf b = *(const v16bf*)&sB[cur][(ni * 16 + l15) * 32 + lhalf * 16];
#pragma unroll
            for (int mi = 0; mi < 2; ++mi)
                acc[mi][ni] = __builtin_amdgcn_wmma_f32_16x16x32_bf16(
                    false, a[mi].v, false, b, (short)0, acc[mi][ni], false, false);
        }
        __syncthreads();
    }

#pragma unroll
    for (int mi = 0; mi < 2; ++mi)
#pragma unroll
    for (int ni = 0; ni < 4; ++ni)
#pragma unroll
    for (int i = 0; i < 8; ++i) {
        int m  = mbase + mi * 16 + i + lhalf * 8;
        int gn = nbase + ni * 16 + l15;
        out[(size_t)m * D_ + gn] = acc[mi][ni][i] + bproj[gn];
    }
}

// ---------------------------------------------------------------------------
// Host launcher
// ---------------------------------------------------------------------------
extern "C" void kernel_launch(void* const* d_in, const int* in_sizes, int n_in,
                              void* d_out, int out_size, void* d_ws, size_t ws_size,
                              hipStream_t stream) {
    const float*         x     = (const float*)d_in[0];
    const float*         bias  = (const float*)d_in[1];
    const unsigned char* kpm   = (const unsigned char*)d_in[2];  // bool mask
    const float*         Wqkv  = (const float*)d_in[3];
    const float*         Wproj = (const float*)d_in[4];
    const float*         bproj = (const float*)d_in[5];
    float*               out   = (float*)d_out;

    __bf16* ws = (__bf16*)d_ws;
    size_t o = 0;
    __bf16* xb  = ws + o; o += (size_t)B_ * N_ * D_;       // x bf16
    __bf16* wqb = ws + o; o += (size_t)3 * D_ * D_;        // Wqkv bf16
    __bf16* wpb = ws + o; o += (size_t)D_ * D_;            // Wproj bf16
    __bf16* qb  = ws + o; o += (size_t)B_ * H_ * N_ * DH_; // Q (x SCALE)
    __bf16* kbf = ws + o; o += (size_t)B_ * H_ * N_ * DH_; // K
    __bf16* vt  = ws + o; o += (size_t)B_ * H_ * DH_ * N_; // V transposed
    __bf16* yw  = ws + o; o += (size_t)B_ * N_ * D_;       // attention out

    const int nx  = B_ * N_ * D_;
    const int nwq = 3 * D_ * D_;
    const int nwp = D_ * D_;
    cvt_f32_bf16<<<(nx  + 255) / 256, 256, 0, stream>>>(x,     xb,  nx);
    cvt_f32_bf16<<<(nwq + 255) / 256, 256, 0, stream>>>(Wqkv,  wqb, nwq);
    cvt_f32_bf16<<<(nwp + 255) / 256, 256, 0, stream>>>(Wproj, wpb, nwp);

    qkv_gemm<<<dim3(B_ * N_ / 128, 3 * D_ / 64), 128, 0, stream>>>(xb, wqb, qb, kbf, vt);

    flash_attn<<<B_ * H_ * (N_ / 16), 32, 0, stream>>>(qb, kbf, vt, bias, kpm, yw);

    proj_gemm<<<dim3(B_ * N_ / 128, D_ / 64), 128, 0, stream>>>(yw, wpb, bproj, out);
}
